// Transformer_35476429865987
// MI455X (gfx1250) — matrix operand order repaired and checked by reading, not compiled, not run on hardware
//
#include <hip/hip_runtime.h>
#include <hip/hip_bf16.h>
#include <math.h>

// ---------------------------------------------------------------------------
// Model dims (compile-time constants matching the reference)
// ---------------------------------------------------------------------------
#define NL    2
#define NH    64
#define SEQ   64
#define DIM   2048
#define HD    32
#define FFN   8192
#define VOCAB 50257
#define BATCH 32
#define ROWS  (BATCH * SEQ)          // 2048 token rows

// CDNA5 async global->LDS path (gated so a missing builtin falls back cleanly)
#if defined(__has_builtin)
#if __has_builtin(__builtin_amdgcn_global_load_async_to_lds_b128) && \
    __has_builtin(__builtin_amdgcn_s_wait_asynccnt)
#define USE_ASYNC 1
#endif
#endif
#ifndef USE_ASYNC
#define USE_ASYNC 0
#endif

typedef __bf16 bf16_t;
typedef __attribute__((ext_vector_type(16))) __bf16 v16bf;
typedef __attribute__((ext_vector_type(8)))  float  v8f;

#if USE_ASYNC
// builtin signature (from hipcc diagnostic): (int4 AS1*, int4 AS3*, imm, imm)
typedef int v4i __attribute__((ext_vector_type(4)));
typedef __attribute__((address_space(1))) v4i gv4i_t;
typedef __attribute__((address_space(3))) v4i lv4i_t;
#endif

struct alignas(16) Half16 { unsigned int d[4]; };
union FragU { v16bf v; Half16 h[2]; };

__device__ inline unsigned short f2bf(float f) {
  union { float f; unsigned int u; } c; c.f = f;
  unsigned int u = c.u;
  unsigned int r = u + 0x7FFFu + ((u >> 16) & 1u);   // round-to-nearest-even
  return (unsigned short)(r >> 16);
}

// A-matrix (16x32 bf16) fragment from row-major 16-bit LDS tile.
// ISA layout: lane<16 -> row=lane, K={0..7,16..23}; lane>=16 -> row=lane-16, K={8..15,24..31}
__device__ inline v16bf load_fragA(const unsigned short* base, int ld, int row_base, int col_base) {
  int lane = threadIdx.x & 31;
  int half = lane >> 4;
  int r = row_base + (lane & 15);
  const unsigned short* p = base + r * ld + col_base + half * 8;
  FragU f;
  f.h[0] = *(const Half16*)(p);        // K +0..7
  f.h[1] = *(const Half16*)(p + 16);   // K +16..23
  return f.v;
}

// B-matrix (32x16 bf16) fragment. Lane n holds B[:,n]; lanes 0-15 K={0..7,16..23},
// lanes 16-31 K={8..15,24..31}. Source tile stored as Bt[n][k] row-major, stride ld.
__device__ inline v16bf load_fragB(const unsigned short* base, int ld, int row_base, int col_base) {
  int lane = threadIdx.x & 31;
  int half = lane >> 4;
  int r = row_base + (lane & 15);
  const unsigned short* p = base + r * ld + col_base + half * 8;
  FragU f;
  f.h[0] = *(const Half16*)(p);        // K +0..7
  f.h[1] = *(const Half16*)(p + 16);   // K +16..23
  return f.v;
}

// ---------------------------------------------------------------------------
// Embedding: x = tok_emb[idx] + sinusoidal pos emb (base 1000, per source)
// ---------------------------------------------------------------------------
__global__ void __launch_bounds__(256)
embed_kernel(const int* __restrict__ idx, const float* __restrict__ tok,
             float* __restrict__ x) {
  int row = blockIdx.x;                 // b*SEQ + s
  int s = row & (SEQ - 1);
  int t = idx[row];
  const float c = -6.90775527898213705f / (float)DIM;   // -ln(1000)/DIM
  for (int j = threadIdx.x; j < DIM; j += 256) {
    int p = j >> 1;
    float ang = (float)s * __expf((float)(2 * p) * c);
    float pe = (j & 1) ? __cosf(ang) : __sinf(ang);
    x[(long)row * DIM + j] = tok[(long)t * DIM + j] + pe;
  }
}

// ---------------------------------------------------------------------------
// Generic GEMM: C[M,N] = act(A[M,K] @ W[K,N] + bias[N] (+ res[M,N]))
// BM=128 BN=128 BK=64. 256 threads = 8 waves (4x2), wave tile 32x64:
// 16 WMMAs per K-stage per wave. f32 tiles arrive via double-buffered
// GLOBAL_LOAD_ASYNC_TO_LDS_B128 (ASYNCcnt pipeline) when available.
// ---------------------------------------------------------------------------
template <bool RELU, bool HASRES>
__global__ void __launch_bounds__(256)
gemm_kernel(const float* __restrict__ A, const float* __restrict__ W,
            const float* __restrict__ bias, const float* __restrict__ res,
            float* __restrict__ C, int M, int N, int K) {
  __shared__ __align__(16) unsigned short A16[128][64];   // bf16 A tile
  __shared__ __align__(16) unsigned short B16[128][64];   // bf16 Bt[n][k]
#if USE_ASYNC
  __shared__ __align__(16) float Af[2][128][64];          // f32 A stage (dbl buf)
  __shared__ __align__(16) float Bf[2][64][128];          // f32 B stage (dbl buf)
#endif
  const int nb = blockIdx.x * 128;
  const int mb = blockIdx.y * 128;
  const int tid  = threadIdx.x;
  const int wave = tid >> 5, lane = tid & 31;
  const int wm = wave >> 1, wn = wave & 1;       // 4 x 2 wave grid
  const int half = lane >> 4, l15 = lane & 15;

  v8f acc[2][4] = {};
  const int nstages = K / 64;

#if USE_ASYNC
  auto issue_stage = [&](int kb, int buf) {
    // A tile: 128 rows x 64 f32 = 2048 16B chunks, 8 per thread
    #pragma unroll
    for (int i = 0; i < 8; ++i) {
      int c = tid + 256 * i;
      int r = c >> 4, cf = (c & 15) * 4;
      const float* src = A + (long)(mb + r) * K + kb + cf;
      __builtin_amdgcn_global_load_async_to_lds_b128(
          (gv4i_t*)(void*)src, (lv4i_t*)(void*)&Af[buf][r][cf], 0, 0);
    }
    // B tile: 64 rows x 128 f32 = 2048 chunks; clamp OOB columns into range
    #pragma unroll
    for (int i = 0; i < 8; ++i) {
      int c = tid + 256 * i;
      int kr = c >> 5, cf = (c & 31) * 4;
      int col = nb + cf;
      if (col > N - 4) col = N - 4;              // clamped chunks hold shifted cols, incl. valid ones if N%4 != 0
      const float* src = W + (long)(kb + kr) * N + col;
      __builtin_amdgcn_global_load_async_to_lds_b128(
          (gv4i_t*)(void*)src, (lv4i_t*)(void*)&Bf[buf][kr][cf], 0, 0);
    }
  };
  issue_stage(0, 0);
#endif

  for (int s = 0; s < nstages; ++s) {
    const int kb = s * 64;
#if USE_ASYNC
    const int buf = s & 1;
    if (s + 1 < nstages) {
      issue_stage(kb + 64, buf ^ 1);             // prefetch next stage
      __builtin_amdgcn_s_wait_asynccnt(16);      // this stage's 16 issues done
    } else {
      __builtin_amdgcn_s_wait_asynccnt(0);
    }
    __syncthreads();   // all waves' async landed; prior frag reads also done
    // convert f32 stage -> bf16 compute tiles (B transposed on the way)
    #pragma unroll
    for (int i = 0; i < 32; ++i) {
      int e = tid + 256 * i;                     // 0..8191
      int r = e >> 6, k = e & 63;
      A16[r][k] = f2bf(Af[buf][r][k]);
    }
    #pragma unroll
    for (int i = 0; i < 32; ++i) {
      int e = tid + 256 * i;
      int k = e >> 7, n = e & 127;
      B16[n][k] = f2bf(Bf[buf][k][n]);
    }
#else
    __syncthreads();   // prior frag reads done before overwrite
    #pragma unroll
    for (int i = 0; i < 32; ++i) {
      int e = tid + 256 * i;
      int r = e >> 6, k = e & 63;
      A16[r][k] = f2bf(A[(long)(mb + r) * K + kb + k]);
    }
    #pragma unroll
    for (int i = 0; i < 32; ++i) {
      int e = tid + 256 * i;
      int k = e >> 7, n = e & 127;
      int col = nb + n;
      float v = (col < N) ? W[(long)(kb + k) * N + col] : 0.f;
      B16[n][k] = f2bf(v);
    }
#endif
    __syncthreads();

    // 2 K-steps x (2 m-tiles x 4 n-tiles) = 16 WMMAs
    #pragma unroll
    for (int kk = 0; kk < 2; ++kk) {
      v16bf a0 = load_fragA(&A16[0][0], 64, 32 * wm,      32 * kk);
      v16bf a1 = load_fragA(&A16[0][0], 64, 32 * wm + 16, 32 * kk);
      #pragma unroll
      for (int j = 0; j < 4; ++j) {
        v16bf b = load_fragB(&B16[0][0], 64, 64 * wn + 16 * j, 32 * kk);
        acc[0][j] = __builtin_amdgcn_wmma_f32_16x16x32_bf16(false, a0, false, b, (short)0, acc[0][j], false, false);
        acc[1][j] = __builtin_amdgcn_wmma_f32_16x16x32_bf16(false, a1, false, b, (short)0, acc[1][j], false, false);
      }
    }
  }

  // Epilogue. C layout: lane<16 -> M=r, lane>=16 -> M=r+8; N = 16*tile + (lane&15)
  #pragma unroll
  for (int mi = 0; mi < 2; ++mi)
    #pragma unroll
    for (int ni = 0; ni < 4; ++ni) {
      int col = nb + 64 * wn + 16 * ni + l15;
      if (col >= N) continue;
      int rowb = mb + 32 * wm + 16 * mi + 8 * half;
      float bv = bias[col];
      #pragma unroll
      for (int r = 0; r < 8; ++r) {
        long o = (long)(rowb + r) * N + col;
        float v = acc[mi][ni][r] + bv;
        if (HASRES) v += res[o];
        if (RELU)   v = fmaxf(v, 0.f);
        C[o] = v;
      }
    }
}

// ---------------------------------------------------------------------------
// Attention per (b,h): scores = Q K^T (unscaled), softmax, AO = P V
// 128 threads = 4 waves; each wave owns 16 rows. LDS-staged bf16 tiles.
// ---------------------------------------------------------------------------
__global__ void __launch_bounds__(128)
attn_kernel(const float* __restrict__ Qr, const float* __restrict__ Kr,
            const float* __restrict__ Vr, float* __restrict__ AO) {
  __shared__ __align__(16) unsigned short Qs[SEQ][HD];
  __shared__ __align__(16) unsigned short Ks[SEQ][HD];
  __shared__ __align__(16) unsigned short Vts[HD][SEQ];   // V transposed: Vt[d][s]
  __shared__ __align__(16) unsigned short Ps[SEQ][SEQ];

  const int bh = blockIdx.x;                 // b*NH + h
  const int b = bh >> 6, h = bh & 63;
  const int tid = threadIdx.x;
  const long base = (long)b * SEQ * DIM + h * HD;   // row s at base + s*DIM

  // stage Q, K (row-major 64x32)
  {
    int s = tid >> 1, hf = tid & 1;
    const float* q = Qr + base + (long)s * DIM + hf * 16;
    const float* k = Kr + base + (long)s * DIM + hf * 16;
    #pragma unroll
    for (int i = 0; i < 16; ++i) {
      Qs[s][hf * 16 + i] = f2bf(q[i]);
      Ks[s][hf * 16 + i] = f2bf(k[i]);
    }
  }
  // stage V transposed (32x64)
  #pragma unroll
  for (int i = 0; i < 16; ++i) {
    int ix = tid * 16 + i;
    int s = ix >> 5, d = ix & 31;
    Vts[d][s] = f2bf(Vr[base + (long)s * DIM + d]);
  }
  __syncthreads();

  const int wave = tid >> 5, lane = tid & 31;
  const int half = lane >> 4, l15 = lane & 15;

  // scores: B-matrix = K^T, so its columns are rows of K -> load_fragB on Ks
  v16bf aq = load_fragA(&Qs[0][0], HD, 16 * wave, 0);
  v8f sc[4];
  #pragma unroll
  for (int n = 0; n < 4; ++n) {
    v16bf bk = load_fragB(&Ks[0][0], HD, 16 * n, 0);
    v8f z = {};
    sc[n] = __builtin_amdgcn_wmma_f32_16x16x32_bf16(false, aq, false, bk, (short)0, z, false, false);
  }

  // row softmax: each row lives in one accumulator index across 16 lanes x 4 frags
  #pragma unroll
  for (int r = 0; r < 8; ++r) {
    float mx = -3.4e38f;
    #pragma unroll
    for (int n = 0; n < 4; ++n) mx = fmaxf(mx, sc[n][r]);
    for (int m = 1; m < 16; m <<= 1) mx = fmaxf(mx, __shfl_xor(mx, m, 32));
    float p[4], sum = 0.f;
    #pragma unroll
    for (int n = 0; n < 4; ++n) { p[n] = __expf(sc[n][r] - mx); sum += p[n]; }
    for (int m = 1; m < 16; m <<= 1) sum += __shfl_xor(sum, m, 32);
    float inv = 1.f / sum;
    int row = 16 * wave + r + 8 * half;
    #pragma unroll
    for (int n = 0; n < 4; ++n) Ps[row][16 * n + l15] = f2bf(p[n] * inv);
  }
  __syncthreads();

  // AO = P(64x64) @ V(64x32): two K-steps, two N tiles
  const long aobase = (long)bh * SEQ * HD;
  #pragma unroll
  for (int ni = 0; ni < 2; ++ni) {
    v8f acc = {};
    #pragma unroll
    for (int t = 0; t < 2; ++t) {
      v16bf pa = load_fragA(&Ps[0][0], SEQ, 16 * wave, 32 * t);
      v16bf vb = load_fragB(&Vts[0][0], SEQ, 16 * ni, 32 * t);
      acc = __builtin_amdgcn_wmma_f32_16x16x32_bf16(false, pa, false, vb, (short)0, acc, false, false);
    }
    int col = 16 * ni + l15;
    int rowb = 16 * wave + 8 * half;
    #pragma unroll
    for (int r = 0; r < 8; ++r)
      AO[aobase + (long)(rowb + r) * HD + col] = acc[r];
  }
}

// ---------------------------------------------------------------------------
// The source's shard/concat/view dance reduces to a pure index permutation:
// out[b,s,j] = AO[b, g=(j/16)%64, i=2*(s%32)+j/1024, e2=(j%16)+16*(s/32)]
// ---------------------------------------------------------------------------
__global__ void __launch_bounds__(256)
scatter_kernel(const float* __restrict__ AO, float* __restrict__ out) {
  int row = blockIdx.x;                 // b*SEQ + s
  int b = row >> 6, s = row & 63;
  int hi = s >> 5, sl = s & 31;
  for (int j = threadIdx.x; j < DIM; j += 256) {
    int g  = (j >> 4) & 63;
    int i  = 2 * sl + (j >> 10);
    int e2 = (j & 15) + 16 * hi;
    out[(long)row * DIM + j] = AO[(((long)b * NH + g) * SEQ + i) * HD + e2];
  }
}

// ---------------------------------------------------------------------------
// LayerNorm over DIM per row, in place (values held in registers)
// ---------------------------------------------------------------------------
__global__ void __launch_bounds__(256)
ln_kernel(float* __restrict__ x, const float* __restrict__ g,
          const float* __restrict__ bta) {
  __shared__ float red[256];
  int row = blockIdx.x;
  float* xr = x + (long)row * DIM;
  float v[8], s = 0.f;
  #pragma unroll
  for (int i = 0; i < 8; ++i) { v[i] = xr[threadIdx.x + 256 * i]; s += v[i]; }
  red[threadIdx.x] = s; __syncthreads();
  for (int off = 128; off > 0; off >>= 1) {
    if (threadIdx.x < off) red[threadIdx.x] += red[threadIdx.x + off];
    __syncthreads();
  }
  float mean = red[0] * (1.f / DIM);
  __syncthreads();
  s = 0.f;
  #pragma unroll
  for (int i = 0; i < 8; ++i) { float d = v[i] - mean; s += d * d; }
  red[threadIdx.x] = s; __syncthreads();
  for (int off = 128; off > 0; off >>= 1) {
    if (threadIdx.x < off) red[threadIdx.x] += red[threadIdx.x + off];
    __syncthreads();
  }
  float rstd = rsqrtf(red[0] * (1.f / DIM) + 1e-5f);
  #pragma unroll
  for (int i = 0; i < 8; ++i) {
    int j = threadIdx.x + 256 * i;
    xr[j] = (v[i] - mean) * rstd * g[j] + bta[j];
  }
}

// ---------------------------------------------------------------------------
// Loss: per-row (lse - logit[target]) then deterministic single-block mean
// ---------------------------------------------------------------------------
__global__ void __launch_bounds__(256)
rowloss_kernel(const float* __restrict__ logits, const int* __restrict__ tgt,
               float* __restrict__ rl) {
  __shared__ float red[256];
  int row = blockIdx.x;
  const float* lr = logits + (long)row * VOCAB;
  float mx = -3.4e38f;
  for (int j = threadIdx.x; j < VOCAB; j += 256) mx = fmaxf(mx, lr[j]);
  red[threadIdx.x] = mx; __syncthreads();
  for (int off = 128; off > 0; off >>= 1) {
    if (threadIdx.x < off) red[threadIdx.x] = fmaxf(red[threadIdx.x], red[threadIdx.x + off]);
    __syncthreads();
  }
  mx = red[0]; __syncthreads();
  float s = 0.f;
  for (int j = threadIdx.x; j < VOCAB; j += 256) s += __expf(lr[j] - mx);
  red[threadIdx.x] = s; __syncthreads();
  for (int off = 128; off > 0; off >>= 1) {
    if (threadIdx.x < off) red[threadIdx.x] += red[threadIdx.x + off];
    __syncthreads();
  }
  if (threadIdx.x == 0) {
    float lse = mx + __logf(red[0]);
    rl[row] = lse - lr[tgt[row]];
  }
}

__global__ void __launch_bounds__(256)
loss_reduce_kernel(const float* __restrict__ rl, float* __restrict__ out) {
  __shared__ float red[256];
  float s = 0.f;
  for (int i = threadIdx.x; i < ROWS; i += 256) s += rl[i];
  red[threadIdx.x] = s; __syncthreads();
  for (int off = 128; off > 0; off >>= 1) {
    if (threadIdx.x < off) red[threadIdx.x] += red[threadIdx.x + off];
    __syncthreads();
  }
  if (threadIdx.x == 0) out[0] = red[0] * (1.f / ROWS);
}

// ---------------------------------------------------------------------------
// Host orchestration
// ---------------------------------------------------------------------------
extern "C" void kernel_launch(void* const* d_in, const int* in_sizes, int n_in,
                              void* d_out, int out_size, void* d_ws, size_t ws_size,
                              hipStream_t stream) {
  const int*   idx     = (const int*)d_in[0];
  const int*   targets = (const int*)d_in[1];
  const float* tok_emb = (const float*)d_in[2];
  const float* Wq  = (const float*)d_in[3];
  const float* bq  = (const float*)d_in[4];
  const float* Wk  = (const float*)d_in[5];
  const float* bk  = (const float*)d_in[6];
  const float* Wv  = (const float*)d_in[7];
  const float* bv  = (const float*)d_in[8];
  const float* Wp  = (const float*)d_in[9];
  const float* bp  = (const float*)d_in[10];
  const float* ln1g = (const float*)d_in[11];
  const float* ln1b = (const float*)d_in[12];
  const float* W1  = (const float*)d_in[13];
  const float* b1  = (const float*)d_in[14];
  const float* W2  = (const float*)d_in[15];
  const float* b2  = (const float*)d_in[16];
  const float* ln2g = (const float*)d_in[17];
  const float* ln2b = (const float*)d_in[18];
  const float* lnfg = (const float*)d_in[19];
  const float* lnfb = (const float*)d_in[20];
  const float* Wlm = (const float*)d_in[21];
  const float* blm = (const float*)d_in[22];

  float* logits = (float*)d_out;                         // [ROWS, VOCAB]
  float* lossp  = logits + (long)ROWS * VOCAB;           // final scalar

  // workspace layout (floats)
  const size_t NX = (size_t)ROWS * DIM;                  // 4,194,304
  float* ws = (float*)d_ws;
  float* X0   = ws;                  // activations (persistent across layers)
  float* X1   = ws + NX;             // attention-block output / FFN residual
  float* Qb   = ws + 2 * NX;         // also reused as PERM, then H1 region
  float* Kb   = ws + 3 * NX;
  float* Vb   = ws + 4 * NX;
  float* AOb  = ws + 5 * NX;
  float* PERM = Qb;                  // q dead after attn
  float* H1   = Qb;                  // spans Qb..AOb (16M floats = [ROWS,FFN])
  float* RL   = X1;                  // row losses at the very end (X1 dead)

  dim3 blk(256);
  dim3 gN2048(DIM / 128, ROWS / 128);         // N=2048 GEMMs
  dim3 gFFN(FFN / 128, ROWS / 128);           // N=8192
  dim3 gLM((VOCAB + 127) / 128, ROWS / 128);  // N=50257

  embed_kernel<<<ROWS, blk, 0, stream>>>(idx, tok_emb, X0);

  for (int l = 0; l < NL; ++l) {
    const float* Wq_l = Wq + (size_t)l * DIM * DIM;
    const float* Wk_l = Wk + (size_t)l * DIM * DIM;
    const float* Wv_l = Wv + (size_t)l * DIM * DIM;
    const float* Wp_l = Wp + (size_t)l * DIM * DIM;
    const float* W1_l = W1 + (size_t)l * DIM * FFN;
    const float* W2_l = W2 + (size_t)l * FFN * DIM;
    const float* bq_l = bq + (size_t)l * DIM;
    const float* bk_l = bk + (size_t)l * DIM;
    const float* bv_l = bv + (size_t)l * DIM;
    const float* bp_l = bp + (size_t)l * DIM;
    const float* b1_l = b1 + (size_t)l * FFN;
    const float* b2_l = b2 + (size_t)l * DIM;

    gemm_kernel<false, false><<<gN2048, blk, 0, stream>>>(X0, Wq_l, bq_l, nullptr, Qb, ROWS, DIM, DIM);
    gemm_kernel<false, false><<<gN2048, blk, 0, stream>>>(X0, Wk_l, bk_l, nullptr, Kb, ROWS, DIM, DIM);
    gemm_kernel<false, false><<<gN2048, blk, 0, stream>>>(X0, Wv_l, bv_l, nullptr, Vb, ROWS, DIM, DIM);

    attn_kernel<<<BATCH * NH, dim3(128), 0, stream>>>(Qb, Kb, Vb, AOb);
    scatter_kernel<<<ROWS, blk, 0, stream>>>(AOb, PERM);

    // x = x + out @ Wp + bp ; LN1 in place
    gemm_kernel<false, true><<<gN2048, blk, 0, stream>>>(PERM, Wp_l, bp_l, X0, X1, ROWS, DIM, DIM);
    ln_kernel<<<ROWS, blk, 0, stream>>>(X1, ln1g + (size_t)l * DIM, ln1b + (size_t)l * DIM);

    // h = relu(x@W1+b1) ; x = LN2(x + h@W2 + b2)
    gemm_kernel<true, false><<<gFFN, blk, 0, stream>>>(X1, W1_l, b1_l, nullptr, H1, ROWS, FFN, DIM);
    gemm_kernel<false, true><<<gN2048, blk, 0, stream>>>(H1, W2_l, b2_l, X1, X0, ROWS, DIM, FFN);
    ln_kernel<<<ROWS, blk, 0, stream>>>(X0, ln2g + (size_t)l * DIM, ln2b + (size_t)l * DIM);
  }

  ln_kernel<<<ROWS, blk, 0, stream>>>(X0, lnfg, lnfb);
  gemm_kernel<false, false><<<gLM, blk, 0, stream>>>(X0, Wlm, blm, nullptr, logits, ROWS, VOCAB, DIM);

  rowloss_kernel<<<ROWS, blk, 0, stream>>>(logits, targets, RL);
  loss_reduce_kernel<<<1, blk, 0, stream>>>(RL, lossp);
}
